// FreqDEBlockSCAM_34445637714146
// MI455X (gfx1250) — compile-verified, matrix operand order checked by reading, not run on hardware
//
#include <hip/hip_runtime.h>

typedef __attribute__((ext_vector_type(16))) _Float16 v16h;
typedef __attribute__((ext_vector_type(8)))  float    v8f;

constexpr int Bn = 2, C = 128, Hh = 96, Ww = 192;
constexpr int HW   = Hh * Ww;        // 18432
constexpr int SZ   = Bn * C * HW;    // 4718592
constexpr int NBHW = Bn * HW;        // 36864
constexpr size_t SMEM_ATTN = 192 * 192 * 4 + 192 * 128 * 2 + 192 * 192 * 2; // 270336 B

// ---------------------------------------------------------------------------
// WMMA helpers (CDNA5 16x16x32 f16 -> f32)
// ---------------------------------------------------------------------------
__device__ __forceinline__ v16h frag_ld(const _Float16* p0, int stride, int lane) {
  // p0 points at [tile_row0][k0] of an LDS tile stored [major][k] (k contiguous).
  // A 16x32 f16 layout: lanes 0-15 -> M=lane, K in {0..7}U{16..23}; lanes 16-31: +8.
  // B 32x16 stored as [N][K] gives the identical per-lane dword-pair pattern.
  const _Float16* p = p0 + (lane & 15) * stride + ((lane & 16) ? 8 : 0);
  v16h f;
#pragma unroll
  for (int v = 0; v < 8; ++v) {
    const int kp = (v < 4) ? (2 * v) : (16 + 2 * (v - 4));
    f[2 * v]     = p[kp];
    f[2 * v + 1] = p[kp + 1];
  }
  return f;
}

__device__ __forceinline__ v8f wmma_f16(v16h a, v16h b, v8f c) {
  return __builtin_amdgcn_wmma_f32_16x16x32_f16(false, a, false, b, (short)0, c,
                                                false, false);
}

__device__ __forceinline__ float gelu_tanh(float x) {
  const float k = 0.7978845608028654f;
  return 0.5f * x * (1.f + tanhf(k * (x + 0.044715f * x * x * x)));
}

// ---------------------------------------------------------------------------
// B-tile loader functors: B(k, n) as float
// ---------------------------------------------------------------------------
struct LdPlain {            // x:[B,Kc,H,W]; n=(b,hw)
  const float* x; int Kc;
  __device__ float operator()(int k, int n) const {
    int b = n / HW, hw = n - b * HW;
    return x[(size_t)(b * Kc + k) * HW + hw];
  }
};
struct LdConcat {           // concat([xa, xb], axis=channel), 128 each
  const float* xa; const float* xb;
  __device__ float operator()(int k, int n) const {
    int b = n / HW, hw = n - b * HW;
    const float* x = (k < C) ? xa : xb;
    int kk = k & (C - 1);
    return x[(size_t)(b * C + kk) * HW + hw];
  }
};
struct LdIm2col {           // 3x3 SAME, x:[B,128,H,W]; k = c*9 + ky*3 + kx
  const float* x;
  __device__ float operator()(int k, int n) const {
    int b = n / HW, hw = n - b * HW;
    int c = k / 9, r = k - c * 9;
    int h = hw / Ww, w = hw - h * Ww;
    int y = h + r / 3 - 1, xx = w + (r - (r / 3) * 3) - 1;
    if ((unsigned)y >= (unsigned)Hh || (unsigned)xx >= (unsigned)Ww) return 0.f;
    return x[(size_t)((b * C + c) * Hh + y) * Ww + xx];
  }
};
struct LdFFT1 {             // pass1: B(k=y', n=(slice, x)) = X[s, y', x]
  const float* x;
  __device__ float operator()(int k, int n) const {
    int s = n / Ww, xc = n - s * Ww;
    return x[(size_t)s * HW + k * Ww + xc];
  }
};
struct LdFFT2 {             // pass2: B(k=x', n=(slice, y)) = tmp[s, y, x']
  const float* t;
  __device__ float operator()(int k, int n) const {
    int s = n / Hh, y = n - s * Hh;
    return t[(size_t)s * HW + y * Ww + k];
  }
};
struct LdDcn {              // bilinear gather of ch via precomputed table
  const float* ch; const float* tab;
  __device__ float operator()(int k, int n) const {
    int b = n / HW, hw = n - b * HW;
    int c = k / 9, kk = k - c * 9;
    const float* t = tab + (size_t)(((b * 9 + kk) * HW) + hw) * 6;
    float w00 = t[0], w01 = t[1], w10 = t[2], w11 = t[3];
    int y0 = __float_as_int(t[4]), x0 = __float_as_int(t[5]);
    int y0c = y0 < 0 ? 0 : (y0 > Hh - 1 ? Hh - 1 : y0);
    int y1t = y0 + 1; int y1c = y1t < 0 ? 0 : (y1t > Hh - 1 ? Hh - 1 : y1t);
    int x0c = x0 < 0 ? 0 : (x0 > Ww - 1 ? Ww - 1 : x0);
    int x1t = x0 + 1; int x1c = x1t < 0 ? 0 : (x1t > Ww - 1 ? Ww - 1 : x1t);
    const float* base = ch + (size_t)(b * C + c) * HW;
    return w00 * base[y0c * Ww + x0c] + w01 * base[y0c * Ww + x1c] +
           w10 * base[y1c * Ww + x0c] + w11 * base[y1c * Ww + x1c];
  }
};

// ---------------------------------------------------------------------------
// Epilogue functors: store(m, n, acc)
// ---------------------------------------------------------------------------
struct EpiBias {            // out:[B,O,H,W] = acc + bias[m]
  float* out; const float* bias; int O;
  __device__ void store(int m, int n, float a) const {
    int b = n / HW, hw = n - b * HW;
    out[(size_t)(b * O + m) * HW + hw] = a + bias[m];
  }
};
struct EpiGelu {
  float* out; const float* bias; int O;
  __device__ void store(int m, int n, float a) const {
    int b = n / HW, hw = n - b * HW;
    out[(size_t)(b * O + m) * HW + hw] = gelu_tanh(a + bias[m]);
  }
};
struct EpiFFT1 {            // tmp[s, m(=y), x]
  float* t;
  __device__ void store(int m, int n, float a) const {
    int s = n / Ww, xc = n - s * Ww;
    t[(size_t)s * HW + m * Ww + xc] = a;
  }
};
struct EpiFFT2 {            // low = acc ; high = orig - acc
  float* low; float* high; const float* orig;
  __device__ void store(int m, int n, float a) const {
    int s = n / Hh, y = n - s * Hh;
    size_t idx = (size_t)s * HW + y * Ww + m;
    low[idx]  = a;
    high[idx] = orig[idx] - a;
  }
};

// ---------------------------------------------------------------------------
// Generic WMMA GEMM: D[M,N] = A[M,K] @ B[K,N], A row-major f32, B via functor.
// Block = 256 threads (8 wave32), tile 128x128, BK=32.
// Register-staged double-buffered f16 LDS staging (1 barrier / k-step) plus a
// global_prefetch_b8 stream two k-blocks ahead on the A matrix.
// ---------------------------------------------------------------------------
template <typename BLoad, typename Epi>
__global__ __launch_bounds__(256)
void gemm_wmma_kernel(const float* __restrict__ A, int M, int N, int K, int lda,
                      BLoad bload, Epi epi) {
  __shared__ _Float16 Ash[2][128 * 32];   // [buf][m][k]
  __shared__ _Float16 Bsh[2][128 * 32];   // [buf][n][k]
  const int tid  = threadIdx.x;
  const int lane = tid & 31;
  const int wave = tid >> 5;
  const int wm = wave >> 2;            // 0..1 (64 rows each)
  const int wn = wave & 3;             // 0..3 (32 cols each)
  const int m0 = blockIdx.y * 128;
  const int n0 = blockIdx.x * 128;
  const int nK = (K + 31) / 32;

  v8f acc[4][2] = {};
  float aR[16], bR[16];

  auto gload = [&](int kb) {
    const int k0 = kb * 32;
#pragma unroll
    for (int j = 0; j < 16; ++j) {
      const int i = tid + j * 256;
      const int mn = i >> 5, k = i & 31;
      const int gk = k0 + k;
      const int gm = m0 + mn;
      const int gn = n0 + mn;
      aR[j] = (gm < M && gk < K) ? A[(size_t)gm * lda + gk] : 0.f;
      bR[j] = (gn < N && gk < K) ? bload(gk, gn) : 0.f;
    }
    // Pull the A stream two k-blocks ahead into cache (global_prefetch_b8).
    const int pm = m0 + (tid >> 1);
    const int pk = k0 + 64 + ((tid & 1) << 4);
    if (pm < M && pk < K) __builtin_prefetch(&A[(size_t)pm * lda + pk], 0, 1);
  };
  auto publish = [&](int buf) {
#pragma unroll
    for (int j = 0; j < 16; ++j) {
      const int i = tid + j * 256;
      Ash[buf][i] = (_Float16)aR[j];
      Bsh[buf][i] = (_Float16)bR[j];
    }
  };

  gload(0);
  publish(0);
  __syncthreads();

  for (int kb = 0; kb < nK; ++kb) {
    const int cur = kb & 1;
    if (kb + 1 < nK) gload(kb + 1);    // overlap next-tile global loads w/ WMMA

    const _Float16* Ab = Ash[cur];
    const _Float16* Bb = Bsh[cur];
    v16h af[4], bf[2];
#pragma unroll
    for (int tm = 0; tm < 4; ++tm)
      af[tm] = frag_ld(Ab + (wm * 64 + tm * 16) * 32, 32, lane);
#pragma unroll
    for (int tn = 0; tn < 2; ++tn)
      bf[tn] = frag_ld(Bb + (wn * 32 + tn * 16) * 32, 32, lane);
#pragma unroll
    for (int tm = 0; tm < 4; ++tm)
#pragma unroll
      for (int tn = 0; tn < 2; ++tn)
        acc[tm][tn] = wmma_f16(af[tm], bf[tn], acc[tm][tn]);

    if (kb + 1 < nK) publish(cur ^ 1); // write ping-pong buffer
    __syncthreads();                   // single barrier per k-step
  }

  const int lm = (lane < 16) ? 0 : 8;
  const int lc = lane & 15;
#pragma unroll
  for (int tm = 0; tm < 4; ++tm)
#pragma unroll
    for (int tn = 0; tn < 2; ++tn)
#pragma unroll
      for (int r = 0; r < 8; ++r) {
        int m = m0 + wm * 64 + tm * 16 + lm + r;
        int n = n0 + wn * 32 + tn * 16 + lc;
        if (m < M && n < N) epi.store(m, n, acc[tm][tn][r]);
      }
}

template <typename BL, typename EP>
static void gemm(const float* A, int M, int N, int K, int lda, BL bl, EP ep,
                 hipStream_t s) {
  dim3 grid((N + 127) / 128, (M + 127) / 128);
  gemm_wmma_kernel<BL, EP><<<grid, dim3(256), 0, s>>>(A, M, N, K, lda, bl, ep);
}

// ---------------------------------------------------------------------------
// Small kernels
// ---------------------------------------------------------------------------
__global__ void init_dirichlet(float* __restrict__ AHm, float* __restrict__ AWm) {
  int i = blockIdx.x * 256 + threadIdx.x;
  if (i < 96 * 96) {
    int d = i / 96 - (i - (i / 96) * 96);
    float s = 1.f;
    for (int u = 1; u <= 24; ++u) s += 2.f * __cosf(6.283185307179586f * u * d / 96.f);
    AHm[i] = s / 96.f;
  }
  if (i < 192 * 192) {
    int d = i / 192 - (i - (i / 192) * 192);
    float s = 1.f;
    for (int u = 1; u <= 48; ++u) s += 2.f * __cosf(6.283185307179586f * u * d / 192.f);
    AWm[i] = s / 192.f;
  }
}

__global__ void dcn_table_kernel(const float* __restrict__ off, float* __restrict__ tab) {
  int i = blockIdx.x * 256 + threadIdx.x;
  if (i >= Bn * 9 * HW) return;
  int hw = i % HW; int t = i / HW; int k = t % 9; int b = t / 9;
  int h = hw / Ww, w = hw - h * Ww;
  float dy = off[(size_t)(b * 18 + 2 * k) * HW + hw];
  float dx = off[(size_t)(b * 18 + 2 * k + 1) * HW + hw];
  float py = (float)h + (float)(k / 3) - 1.f + dy;
  float px = (float)w + (float)(k - (k / 3) * 3) - 1.f + dx;
  float y0 = floorf(py), x0 = floorf(px);
  float fy = py - y0, fx = px - x0;
  auto valid = [](float yy, float xx) {
    return (yy >= 0.f && yy <= (float)(Hh - 1) && xx >= 0.f && xx <= (float)(Ww - 1)) ? 1.f : 0.f;
  };
  float* o = tab + (size_t)i * 6;
  o[0] = (1.f - fy) * (1.f - fx) * valid(y0, x0);
  o[1] = (1.f - fy) * fx * valid(y0, x0 + 1.f);
  o[2] = fy * (1.f - fx) * valid(y0 + 1.f, x0);
  o[3] = fy * fx * valid(y0 + 1.f, x0 + 1.f);
  o[4] = __int_as_float((int)y0);
  o[5] = __int_as_float((int)x0);
}

__global__ void ln_kernel(const float* __restrict__ x, const float* __restrict__ g,
                          const float* __restrict__ bb, float* __restrict__ out) {
  int i = blockIdx.x * 256 + threadIdx.x;
  if (i >= NBHW) return;
  int b = i / HW, hw = i - b * HW;
  const float* p = x + (size_t)b * C * HW + hw;
  float s = 0.f, s2 = 0.f;
  for (int c = 0; c < C; ++c) { float v = p[(size_t)c * HW]; s += v; s2 += v * v; }
  float mu = s / (float)C;
  float var = s2 / (float)C - mu * mu;
  var = var < 0.f ? 0.f : var;
  float inv = rsqrtf(var + 1e-6f);
  float* q = out + (size_t)b * C * HW + hw;
  for (int c = 0; c < C; ++c)
    q[(size_t)c * HW] = (p[(size_t)c * HW] - mu) * inv * g[c] + bb[c];
}

__global__ void high_update_kernel(float* __restrict__ hi, const float* __restrict__ proj,
                                   const float* __restrict__ chx, const float* __restrict__ bsc) {
  int i = blockIdx.x * 256 + threadIdx.x;
  if (i >= SZ) return;
  int c = (i / HW) & (C - 1);
  hi[i] += proj[i] * chx[i] * bsc[c];
}

__global__ void final_add_kernel(const float* __restrict__ lh, const float* __restrict__ ll,
                                 const float* __restrict__ rh, const float* __restrict__ rl,
                                 float* __restrict__ out) {
  int i = blockIdx.x * 256 + threadIdx.x;
  if (i >= SZ) return;
  out[i]      = lh[i] + ll[i];
  out[SZ + i] = rh[i] + rl[i];
}

// ---------------------------------------------------------------------------
// Fused SCAM attention: one workgroup per (b, h) row. 270 KB dynamic LDS
// (CDNA5: up to 320 KB/workgroup). att = Ql^T Qr * scale (192x192, K=128);
// row/col softmax in LDS; two WMMA applies accumulate into the low residuals.
// ---------------------------------------------------------------------------
__global__ __launch_bounds__(256)
void scam_attn_kernel(const float* __restrict__ Ql, const float* __restrict__ Qr,
                      const float* __restrict__ Vl, const float* __restrict__ Vr,
                      float* __restrict__ lLow, float* __restrict__ rLow,
                      const float* __restrict__ sbeta, const float* __restrict__ sgamma) {
  extern __shared__ char smem[];
  float*    att  = (float*)smem;                                         // 192*192 f32
  _Float16* Areg = (_Float16*)(smem + 192 * 192 * 4);                    // 192*128 f16
  _Float16* Sreg = (_Float16*)(smem + 192 * 192 * 4 + 192 * 128 * 2);    // 192*192 f16
  const int bh = blockIdx.x;
  const int b = bh / Hh, h = bh - b * Hh;
  const size_t base = (size_t)b * C * HW + (size_t)h * Ww;  // element (c,w) at +c*HW+w
  const int tid = threadIdx.x, lane = tid & 31, wave = tid >> 5;
  const float scale = 0.08838834764831845f;  // 128^-0.5

  // Stage A = Ql^T  [w][c], B = Qr [v][c]
  for (int i = tid; i < 192 * 128; i += 256) {
    int m = i >> 7, k = i & 127;
    Areg[i] = (_Float16)Ql[base + (size_t)k * HW + m];
    Sreg[i] = (_Float16)Qr[base + (size_t)k * HW + m];
  }
  __syncthreads();

  for (int t = wave; t < 144; t += 8) {
    int tm = t / 12, tn = t - (t / 12) * 12;
    v8f acc = {};
#pragma unroll
    for (int k0 = 0; k0 < 128; k0 += 32)
      acc = wmma_f16(frag_ld(Areg + tm * 16 * 128 + k0, 128, lane),
                     frag_ld(Sreg + tn * 16 * 128 + k0, 128, lane), acc);
    const int lm = (lane < 16) ? 0 : 8, lc = lane & 15;
#pragma unroll
    for (int r = 0; r < 8; ++r)
      att[(tm * 16 + lm + r) * 192 + tn * 16 + lc] = acc[r] * scale;
  }
  __syncthreads();

  // softmax over v (rows) -> S1[w][v] f16 (serves as B in [n=w][k=v] layout)
  if (tid < 192) {
    const int wr = tid;
    float mx = -3.4e38f;
    for (int v = 0; v < 192; ++v) mx = fmaxf(mx, att[wr * 192 + v]);
    float sum = 0.f;
    for (int v = 0; v < 192; ++v) {
      float e = __expf(att[wr * 192 + v] - mx);
      sum += e;
      Sreg[wr * 192 + v] = (_Float16)e;
    }
    float inv = 1.f / sum;
    for (int v = 0; v < 192; ++v)
      Sreg[wr * 192 + v] = (_Float16)((float)Sreg[wr * 192 + v] * inv);
  }
  __syncthreads();

  // A = Vr [c][v]
  for (int i = tid; i < 128 * 192; i += 256) {
    int m = i / 192, k = i - (i / 192) * 192;
    Areg[i] = (_Float16)Vr[base + (size_t)m * HW + k];
  }
  __syncthreads();

  // Fr2l = Vr @ S1^T -> (c, w); lLow += Fr2l * sbeta[c]
  for (int t = wave; t < 96; t += 8) {
    int tm = t / 12, tn = t - (t / 12) * 12;
    v8f acc = {};
#pragma unroll
    for (int k0 = 0; k0 < 192; k0 += 32)
      acc = wmma_f16(frag_ld(Areg + tm * 16 * 192 + k0, 192, lane),
                     frag_ld(Sreg + tn * 16 * 192 + k0, 192, lane), acc);
    const int lm = (lane < 16) ? 0 : 8, lc = lane & 15;
#pragma unroll
    for (int r = 0; r < 8; ++r) {
      int c = tm * 16 + lm + r, wcol = tn * 16 + lc;
      lLow[base + (size_t)c * HW + wcol] += acc[r] * sbeta[c];
    }
  }
  __syncthreads();

  // softmax over w (cols) -> S2 stored transposed [v][w] f16 (B in [n=v][k=w])
  if (tid < 192) {
    const int v = tid;
    float mx = -3.4e38f;
    for (int wr = 0; wr < 192; ++wr) mx = fmaxf(mx, att[wr * 192 + v]);
    float sum = 0.f;
    for (int wr = 0; wr < 192; ++wr) {
      float e = __expf(att[wr * 192 + v] - mx);
      sum += e;
      Sreg[v * 192 + wr] = (_Float16)e;
    }
    float inv = 1.f / sum;
    for (int wr = 0; wr < 192; ++wr)
      Sreg[v * 192 + wr] = (_Float16)((float)Sreg[v * 192 + wr] * inv);
  }
  __syncthreads();

  // A = Vl [c][w]
  for (int i = tid; i < 128 * 192; i += 256) {
    int m = i / 192, k = i - (i / 192) * 192;
    Areg[i] = (_Float16)Vl[base + (size_t)m * HW + k];
  }
  __syncthreads();

  // Fl2r = Vl @ S2 -> (c, v); rLow += Fl2r * sgamma[c]
  for (int t = wave; t < 96; t += 8) {
    int tm = t / 12, tn = t - (t / 12) * 12;
    v8f acc = {};
#pragma unroll
    for (int k0 = 0; k0 < 192; k0 += 32)
      acc = wmma_f16(frag_ld(Areg + tm * 16 * 192 + k0, 192, lane),
                     frag_ld(Sreg + tn * 16 * 192 + k0, 192, lane), acc);
    const int lm = (lane < 16) ? 0 : 8, lc = lane & 15;
#pragma unroll
    for (int r = 0; r < 8; ++r) {
      int c = tm * 16 + lm + r, vcol = tn * 16 + lc;
      rLow[base + (size_t)c * HW + vcol] += acc[r] * sgamma[c];
    }
  }
}

// ---------------------------------------------------------------------------
extern "C" void kernel_launch(void* const* d_in, const int* in_sizes, int n_in,
                              void* d_out, int out_size, void* d_ws, size_t ws_size,
                              hipStream_t stream) {
  (void)in_sizes; (void)n_in; (void)out_size; (void)ws_size;
  const float* inL  = (const float*)d_in[0];
  const float* inR  = (const float*)d_in[1];
  const float* cdw  = (const float*)d_in[2];
  const float* cdb  = (const float*)d_in[3];
  const float* l1w  = (const float*)d_in[4];
  const float* l1b  = (const float*)d_in[5];
  const float* l2w  = (const float*)d_in[6];
  const float* l2b  = (const float*)d_in[7];
  const float* olw  = (const float*)d_in[8];
  const float* olb  = (const float*)d_in[9];
  const float* orw  = (const float*)d_in[10];
  const float* orb  = (const float*)d_in[11];
  const float* dlw  = (const float*)d_in[12];
  const float* dlb  = (const float*)d_in[13];
  const float* drw  = (const float*)d_in[14];
  const float* drb  = (const float*)d_in[15];
  const float* plw  = (const float*)d_in[16];
  const float* plb  = (const float*)d_in[17];
  const float* prw  = (const float*)d_in[18];
  const float* prb  = (const float*)d_in[19];
  const float* beta = (const float*)d_in[20];
  const float* gamm = (const float*)d_in[21];
  const float* lnlg = (const float*)d_in[22];
  const float* lnlb = (const float*)d_in[23];
  const float* lnrg = (const float*)d_in[24];
  const float* lnrb = (const float*)d_in[25];
  const float* lp1w = (const float*)d_in[26];
  const float* lp1b = (const float*)d_in[27];
  const float* rp1w = (const float*)d_in[28];
  const float* rp1b = (const float*)d_in[29];
  const float* lp2w = (const float*)d_in[30];
  const float* lp2b = (const float*)d_in[31];
  const float* rp2w = (const float*)d_in[32];
  const float* rp2b = (const float*)d_in[33];
  const float* sbet = (const float*)d_in[34];
  const float* sgam = (const float*)d_in[35];

  float* wsp = (float*)d_ws;
  auto take = [&](size_t n) { float* r = wsp; wsp += n; return r; };
  float* AHm  = take(96 * 96);
  float* AWm  = take(192 * 192);
  float* tmp  = take(SZ);
  float* lLow = take(SZ); float* lHigh = take(SZ);
  float* rLow = take(SZ); float* rHigh = take(SZ);
  float* ch   = take(SZ); float* t1    = take(SZ);
  float* offL = take((size_t)Bn * 18 * HW);
  float* offR = take((size_t)Bn * 18 * HW);
  float* tabl = take((size_t)Bn * 9 * HW * 6);
  float* chl  = take(SZ); float* chr  = take(SZ);
  float* prjL = take(SZ); float* prjR = take(SZ);
  float* lnL  = take(SZ); float* lnR  = take(SZ);
  float* Qlb  = take(SZ); float* Qrb  = take(SZ);
  float* Vlb  = take(SZ); float* Vrb  = take(SZ);

  init_dirichlet<<<dim3(144), dim3(256), 0, stream>>>(AHm, AWm);

  // FFT low/high split == separable Dirichlet matmuls (exact)
  gemm(AHm, 96, Bn * C * Ww, 96, 96, LdFFT1{inL}, EpiFFT1{tmp}, stream);
  gemm(AWm, 192, Bn * C * Hh, 192, 192, LdFFT2{tmp}, EpiFFT2{lLow, lHigh, inL}, stream);
  gemm(AHm, 96, Bn * C * Ww, 96, 96, LdFFT1{inR}, EpiFFT1{tmp}, stream);
  gemm(AWm, 192, Bn * C * Hh, 192, 192, LdFFT2{tmp}, EpiFFT2{rLow, rHigh, inR}, stream);

  // conv_down (1x1 over concat) -> ch ; lfe1(3x3)+gelu -> t1 ; lfe2(3x3) -> ch
  gemm(cdw, C, NBHW, 2 * C, 2 * C, LdConcat{lHigh, rHigh}, EpiBias{ch, cdb, C}, stream);
  gemm(l1w, C, NBHW, C * 9, C * 9, LdIm2col{ch}, EpiGelu{t1, l1b, C}, stream);
  gemm(l2w, C, NBHW, C * 9, C * 9, LdIm2col{t1}, EpiBias{ch, l2b, C}, stream);

  // offset convs (3x3, O=18)
  gemm(olw, 18, NBHW, C * 9, C * 9, LdIm2col{lHigh}, EpiBias{offL, olb, 18}, stream);
  gemm(orw, 18, NBHW, C * 9, C * 9, LdIm2col{rHigh}, EpiBias{offR, orb, 18}, stream);

  // deformable convs: bilinear table then gather-GEMM (table reused L->R)
  dcn_table_kernel<<<dim3((Bn * 9 * HW + 255) / 256), dim3(256), 0, stream>>>(offL, tabl);
  gemm(dlw, C, NBHW, C * 9, C * 9, LdDcn{ch, tabl}, EpiBias{chl, dlb, C}, stream);
  dcn_table_kernel<<<dim3((Bn * 9 * HW + 255) / 256), dim3(256), 0, stream>>>(offR, tabl);
  gemm(drw, C, NBHW, C * 9, C * 9, LdDcn{ch, tabl}, EpiBias{chr, drb, C}, stream);

  // projections (from pre-update highs), then fused residual update
  gemm(plw, C, NBHW, C, C, LdPlain{lHigh, C}, EpiBias{prjL, plb, C}, stream);
  gemm(prw, C, NBHW, C, C, LdPlain{rHigh, C}, EpiBias{prjR, prb, C}, stream);
  high_update_kernel<<<dim3((SZ + 255) / 256), dim3(256), 0, stream>>>(lHigh, prjL, chl, beta);
  high_update_kernel<<<dim3((SZ + 255) / 256), dim3(256), 0, stream>>>(rHigh, prjR, chr, gamm);

  // SCAM: LN -> Q/V projections -> fused per-(b,h) attention
  ln_kernel<<<dim3((NBHW + 255) / 256), dim3(256), 0, stream>>>(lLow, lnlg, lnlb, lnL);
  ln_kernel<<<dim3((NBHW + 255) / 256), dim3(256), 0, stream>>>(rLow, lnrg, lnrb, lnR);
  gemm(lp1w, C, NBHW, C, C, LdPlain{lnL, C}, EpiBias{Qlb, lp1b, C}, stream);
  gemm(rp1w, C, NBHW, C, C, LdPlain{lnR, C}, EpiBias{Qrb, rp1b, C}, stream);
  gemm(lp2w, C, NBHW, C, C, LdPlain{lLow, C}, EpiBias{Vlb, lp2b, C}, stream);
  gemm(rp2w, C, NBHW, C, C, LdPlain{rLow, C}, EpiBias{Vrb, rp2b, C}, stream);
  scam_attn_kernel<<<dim3(Bn * Hh), dim3(256), SMEM_ATTN, stream>>>(
      Qlb, Qrb, Vlb, Vrb, lLow, rLow, sbet, sgam);

  // outputs: (l_high + l_low, r_high + r_low)
  final_add_kernel<<<dim3((SZ + 255) / 256), dim3(256), 0, stream>>>(
      lHigh, lLow, rHigh, rLow, (float*)d_out);
}
